// GolayConv2d_10874857193723
// MI455X (gfx1250) — matrix operand order repaired and checked by reading, not compiled, not run on hardware
//
#include <hip/hip_runtime.h>
#include <hip/hip_bf16.h>

typedef __attribute__((ext_vector_type(16))) _Float16 v16h;
typedef __attribute__((ext_vector_type(8)))  float    v8f;

#define IN_C   256
#define OUT_C  256
#define HWSZ   3136     // 56*56
#define WD     56
#define KTOT   2304     // 9*256
#define NSTEP  72       // 2304/32

#if defined(__has_builtin)
# if __has_builtin(__builtin_amdgcn_global_load_async_to_lds_b128)
#  define HAVE_ASYNC_LDS 1
# endif
# if __has_builtin(__builtin_amdgcn_s_wait_asynccnt)
#  define HAVE_WAIT_ASYNC 1
# endif
#endif

#if defined(HAVE_ASYNC_LDS)
typedef int async_i4 __attribute__((vector_size(16)));
typedef __attribute__((address_space(1))) async_i4* gp_i4;
typedef __attribute__((address_space(3))) async_i4* lp_i4;

__device__ __forceinline__ void async_ld_b128(const _Float16* g, _Float16* l) {
    __builtin_amdgcn_global_load_async_to_lds_b128(
        (gp_i4)g, (lp_i4)l, 0, 0);
}
#endif

__device__ __forceinline__ void wait_asynccnt0() {
#if defined(HAVE_WAIT_ASYNC)
    __builtin_amdgcn_s_wait_asynccnt(0);
#else
    asm volatile("s_wait_asynccnt 0x0" ::: "memory");
#endif
}

// ---------------------------------------------------------------------------
// Stage 1: golay * x, FWHT-256 over channels, RMS norm, -> f16 NHWC
// One wave handles one pixel (8 channels per lane). 32 pixels per workgroup,
// staged through LDS for coalesced NCHW reads. 3136 % 32 == 0 so a tile never
// crosses an image boundary.
// golay_sequence(n) == Golay-Rudin-Shapiro: g(c) = (-1)^popcount(c & (c>>1))
// ---------------------------------------------------------------------------
__global__ __launch_bounds__(256) void golay_fwht_rms_kernel(
    const float* __restrict__ x, const float* __restrict__ rms_w,
    _Float16* __restrict__ x16)
{
    __shared__ float xs[256 * 33];   // [c][pixel] with pad
    const int tid  = threadIdx.x;
    const int lane = tid & 31;
    const int wv   = tid >> 5;

    const long long p0  = (long long)blockIdx.x * 32;   // first pixel of tile
    const int n   = (int)(p0 / HWSZ);
    const int rp0 = (int)(p0 % HWSZ);
    const float* xbase = x + (long long)n * IN_C * HWSZ + rp0;

    // coalesced load: wave wv loads channels [wv*32, wv*32+32), 32 pixels wide
    for (int j = 0; j < 32; ++j) {
        int c = wv * 32 + j;
        xs[c * 33 + lane] = xbase[(long long)c * HWSZ + lane];
    }
    __syncthreads();

    // each wave processes 4 pixels
    for (int it = 0; it < 4; ++it) {
        const int q = wv * 4 + it;
        float v[8];
        #pragma unroll
        for (int r = 0; r < 8; ++r) {
            int c = lane * 8 + r;
            float g = (__popc(c & (c >> 1)) & 1) ? -1.0f : 1.0f;
            v[r] = xs[c * 33 + q] * g;
        }
        // FWHT strides 1,2,4: in-register
        #pragma unroll
        for (int s = 0; s < 3; ++s) {
            int m = 1 << s;
            #pragma unroll
            for (int r = 0; r < 8; ++r) {
                if (!(r & m)) {
                    float a = v[r], c2 = v[r | m];
                    v[r]     = a + c2;
                    v[r | m] = a - c2;
                }
            }
        }
        // FWHT strides 8..128: cross-lane xor 1..16 (wave32)
        #pragma unroll
        for (int s = 0; s < 5; ++s) {
            int m = 1 << s;
            bool hb = (lane & m) != 0;
            #pragma unroll
            for (int r = 0; r < 8; ++r) {
                float pv = __shfl_xor(v[r], m, 32);
                v[r] = hb ? (pv - v[r]) : (v[r] + pv);
            }
        }
        // RMS over 256 channels
        float ss = 0.0f;
        #pragma unroll
        for (int r = 0; r < 8; ++r) ss += v[r] * v[r];
        #pragma unroll
        for (int m = 1; m <= 16; m <<= 1) ss += __shfl_xor(ss, m, 32);
        float scale = rsqrtf(ss * (1.0f / 256.0f) + 1e-5f);

        union { _Float16 h[8]; int4 q4; } o;
        #pragma unroll
        for (int r = 0; r < 8; ++r) {
            int c = lane * 8 + r;
            o.h[r] = (_Float16)(v[r] * scale * rms_w[c]);
        }
        long long p = p0 + q;
        *(int4*)(x16 + p * IN_C + lane * 8) = o.q4;   // 16B aligned
    }
}

// ---------------------------------------------------------------------------
// Stage 2: W[oc][ic][kh][kw] f32  ->  Wc[oc][kh*3+kw][ic] f16 (K-contiguous)
// ---------------------------------------------------------------------------
__global__ __launch_bounds__(256) void reorder_w_kernel(
    const float* __restrict__ W, _Float16* __restrict__ Wc)
{
    int i   = blockIdx.x * 256 + threadIdx.x;   // over 256*2304
    int oc  = i / KTOT;
    int rem = i - oc * KTOT;
    int k9  = rem >> 8;          // kh*3+kw
    int ic  = rem & 255;
    Wc[i] = (_Float16)W[(oc * 256 + ic) * 9 + k9];
}

// ---------------------------------------------------------------------------
// Stage 3: implicit-GEMM 3x3 conv with v_wmma_f32_16x16x32_f16.
// M = pixels (100352), N = oc (256), K = 2304.  WG = 256 thr = 8 waves,
// tile 128(M) x 128(N); wave = 32x64 (2x4 wmma tiles). B double-buffered in
// LDS via async global->LDS loads; A loaded per-lane from NHWC f16 with
// zero-fill at padding.
// ---------------------------------------------------------------------------
__global__ __launch_bounds__(256) void conv_wmma_kernel(
    const _Float16* __restrict__ x16, const _Float16* __restrict__ Wc,
    const float* __restrict__ bias, float* __restrict__ out)
{
    // rows padded to 40 halfs (80B): keeps 16B alignment, spreads banks
    __shared__ _Float16 Bs[2 * 128 * 40];

    const int tid  = threadIdx.x;
    const int lane = tid & 31;
    const int wv   = tid >> 5;
    const int mw   = wv & 3;      // M quadrant (32 rows each)
    const int nw   = wv >> 2;     // N half (64 cols each)
    const int hi   = lane >> 4;   // K-half selector in fragments
    const int l16  = lane & 15;

    const long long bm0 = (long long)blockIdx.x * 128;
    const int       bn0 = blockIdx.y * 128;

    // per-lane A-row coordinates (step-invariant). 16-row subtiles never
    // cross an image boundary (3136 % 16 == 0).
    long long rowp[2]; int rowh[2], roww[2];
    #pragma unroll
    for (int mi = 0; mi < 2; ++mi) {
        long long p = bm0 + mw * 32 + mi * 16 + l16;
        rowp[mi] = p;
        int rp = (int)(p % HWSZ);
        rowh[mi] = rp / WD;
        roww[mi] = rp % WD;
    }

    v8f acc[2][4];
    const v8f zacc = {0.f, 0.f, 0.f, 0.f, 0.f, 0.f, 0.f, 0.f};
    #pragma unroll
    for (int mi = 0; mi < 2; ++mi)
        #pragma unroll
        for (int ni = 0; ni < 4; ++ni) acc[mi][ni] = zacc;

    // B-tile load mapping: thread -> (oc row, 16-half chunk)
    const _Float16* wsrc = Wc + (long long)(bn0 + (tid >> 1)) * KTOT + (tid & 1) * 16;
    const int bofs = (tid >> 1) * 40 + (tid & 1) * 16;

    // prologue: stage step 0 into buffer 0
#if defined(HAVE_ASYNC_LDS)
    async_ld_b128(wsrc,     Bs + bofs);
    async_ld_b128(wsrc + 8, Bs + bofs + 8);
    wait_asynccnt0();
#else
    {
        int4 b0 = *(const int4*)(wsrc);
        int4 b1 = *(const int4*)(wsrc + 8);
        *(int4*)(Bs + bofs)     = b0;
        *(int4*)(Bs + bofs + 8) = b1;
    }
#endif
    __syncthreads();

    union Frag { v16h v; int4 q[2]; };

    for (int step = 0; step < NSTEP; ++step) {
        const int k256 = step >> 3;          // (step*32)/256
        const int kh   = k256 / 3;
        const int kw   = k256 - kh * 3;
        const int ic0  = (step * 32) & 255;
        const int bufoff = (step & 1) * (128 * 40);

        // stage next B tile into the other buffer
#if defined(HAVE_ASYNC_LDS)
        if (step + 1 < NSTEP) {
            const _Float16* ws = wsrc + (step + 1) * 32;
            _Float16* d = Bs + ((step + 1) & 1) * (128 * 40) + bofs;
            async_ld_b128(ws,     d);
            async_ld_b128(ws + 8, d + 8);
        }
#else
        int4 nb0, nb1;
        if (step + 1 < NSTEP) {
            const _Float16* ws = wsrc + (step + 1) * 32;
            nb0 = *(const int4*)(ws);
            nb1 = *(const int4*)(ws + 8);
        }
#endif

        // A fragments (global, NHWC f16; per-lane two contiguous 16B chunks)
        Frag af[2];
        #pragma unroll
        for (int mi = 0; mi < 2; ++mi) {
            int hh = rowh[mi] + kh - 1;
            int ww = roww[mi] + kw - 1;
            bool valid = ((unsigned)hh < 56u) && ((unsigned)ww < 56u);
            if (valid) {
                const _Float16* src =
                    x16 + (rowp[mi] + (long long)((kh - 1) * WD + (kw - 1))) * IN_C
                        + ic0 + hi * 8;
                af[mi].q[0] = *(const int4*)(src);        // K c0..c0+7
                af[mi].q[1] = *(const int4*)(src + 16);   // K c0+16..c0+23
                __builtin_prefetch(src + 64, 0, 0);       // speculative, next line
            } else {
                af[mi].q[0] = make_int4(0, 0, 0, 0);
                af[mi].q[1] = make_int4(0, 0, 0, 0);
            }
        }

        // B fragments from LDS (col = l16, K hi*8..+7 and 16+hi*8..+7 per lane)
        Frag bf[4];
        #pragma unroll
        for (int ni = 0; ni < 4; ++ni) {
            const _Float16* src =
                Bs + bufoff + (nw * 64 + ni * 16 + l16) * 40 + hi * 8;
            bf[ni].q[0] = *(const int4*)(src);
            bf[ni].q[1] = *(const int4*)(src + 16);
        }

        // 8 WMMAs
        #pragma unroll
        for (int mi = 0; mi < 2; ++mi)
            #pragma unroll
            for (int ni = 0; ni < 4; ++ni)
                acc[mi][ni] = __builtin_amdgcn_wmma_f32_16x16x32_f16(
                    false, af[mi].v, false, bf[ni].v,
                    (short)0, acc[mi][ni], false, false);

#if defined(HAVE_ASYNC_LDS)
        wait_asynccnt0();       // own async writes to buf^1 landed
#else
        if (step + 1 < NSTEP) {
            _Float16* d = Bs + ((step + 1) & 1) * (128 * 40) + bofs;
            *(int4*)(d)     = nb0;
            *(int4*)(d + 8) = nb1;
        }
#endif
        __syncthreads();
    }

    // epilogue: C layout -> out[n][oc][h*56+w] + bias, two b128 stores/tile
    #pragma unroll
    for (int mi = 0; mi < 2; ++mi) {
        long long pb = bm0 + mw * 32 + mi * 16;     // subtile base pixel
        int nimg = (int)(pb / HWSZ);
        int rp0  = (int)(pb % HWSZ);
        #pragma unroll
        for (int ni = 0; ni < 4; ++ni) {
            int oc = bn0 + nw * 64 + ni * 16 + l16;
            float bv = bias[oc];
            float* obase = out + ((long long)nimg * OUT_C + oc) * HWSZ + rp0 + hi * 8;
            float4 o0 = make_float4(acc[mi][ni][0] + bv, acc[mi][ni][1] + bv,
                                    acc[mi][ni][2] + bv, acc[mi][ni][3] + bv);
            float4 o1 = make_float4(acc[mi][ni][4] + bv, acc[mi][ni][5] + bv,
                                    acc[mi][ni][6] + bv, acc[mi][ni][7] + bv);
            *(float4*)(obase)     = o0;   // 16B aligned (rp0 % 16 == 0)
            *(float4*)(obase + 4) = o1;
        }
    }
}

// ---------------------------------------------------------------------------
// Workspace layout: [x16: 100352*256 f16 = 51,380,224 B][Wc: 256*2304 f16]
// Total ~52.6 MB (fully rewritten every launch; 0xAA poison is harmless).
// ---------------------------------------------------------------------------
extern "C" void kernel_launch(void* const* d_in, const int* in_sizes, int n_in,
                              void* d_out, int out_size, void* d_ws, size_t ws_size,
                              hipStream_t stream) {
    const float* x     = (const float*)d_in[0];
    const float* W     = (const float*)d_in[1];
    const float* b     = (const float*)d_in[2];
    const float* rms_w = (const float*)d_in[3];
    float* out = (float*)d_out;

    _Float16* x16 = (_Float16*)d_ws;
    _Float16* Wc  = (_Float16*)((char*)d_ws + (size_t)100352 * 256 * 2);

    golay_fwht_rms_kernel<<<3136, 256, 0, stream>>>(x, rms_w, x16);
    reorder_w_kernel<<<(256 * KTOT) / 256, 256, 0, stream>>>(W, Wc);
    conv_wmma_kernel<<<dim3(784, 2), 256, 0, stream>>>(x16, Wc, b, out);
}
